// MoeBlock_RS_54589034332239
// MI455X (gfx1250) — compile-verified, matrix operand order checked by reading, not run on hardware
//
#include <hip/hip_runtime.h>

// MoE top-2 SwiGLU block for MI455X (gfx1250), bf16 WMMA with f32 accumulate.
// T=4096 tokens, H=2048, F=1280, E=12 experts, TOP_K=2.
// Double-buffered K-pipeline: async global->LDS for activations (ASYNCcnt),
// register-staged fp32->bf16 conversion for weights, one barrier per K-step.

#define Tn 4096
#define Hn 2048
#define Fn 1280
#define En 12
#define LSTRIDE 72   // LDS row stride in bf16 units (144B: 16B-aligned, bank-conflict-free)

typedef __attribute__((ext_vector_type(16))) __bf16 v16bf;
typedef __attribute__((ext_vector_type(8)))  float  v8f;
typedef __attribute__((ext_vector_type(4)))  unsigned int uivec4;

union FragAB { v16bf v; unsigned int u[8]; };
union FragC  { v8f   v; float        f[8]; };

// fp32 -> bf16, round-half-up (2 VALU ops)
__device__ __forceinline__ unsigned short f2bf_fast(float f) {
  union { float f; unsigned int u; } cv; cv.f = f;
  return (unsigned short)((cv.u + 0x8000u) >> 16);
}

// packed fp32x2 -> bf16x2: two biased adds + one v_perm_b32
__device__ __forceinline__ unsigned int pack_bf16(float lo, float hi) {
  union { float f; unsigned int u; } a, b; a.f = lo; b.f = hi;
  return __builtin_amdgcn_perm(b.u + 0x8000u, a.u + 0x8000u, 0x07060302u);
}

// async global -> LDS, 16 bytes per lane (ASYNCcnt-tracked, bypasses VGPRs)
__device__ __forceinline__ void async_copy_b128(unsigned int lds_byte_off,
                                                const void* gaddr) {
  asm volatile("global_load_async_to_lds_b128 %0, %1, off"
               :: "v"(lds_byte_off), "v"((unsigned long long)(size_t)gaddr)
               : "memory");
}
__device__ __forceinline__ void async_wait0() {
  asm volatile("s_wait_asynccnt 0x0" ::: "memory");
}

// ---------------------------------------------------------------------------
// Kernel 1: gating. One wave32 per token.
// ---------------------------------------------------------------------------
__global__ __launch_bounds__(256) void gate_kernel(
    const float* __restrict__ x, const float* __restrict__ gw,
    float* __restrict__ cw, int* __restrict__ cnt, int* __restrict__ idx) {
  const int lane = threadIdx.x & 31;
  const int wv   = threadIdx.x >> 5;
  const int t    = blockIdx.x * 8 + wv;
  if (t >= Tn) return;

  float acc[En];
#pragma unroll
  for (int e = 0; e < En; ++e) acc[e] = 0.0f;

  const float* xr = x + (size_t)t * Hn;
  for (int h = lane; h < Hn; h += 32) {
    float xv = xr[h];
#pragma unroll
    for (int e = 0; e < En; ++e) acc[e] += xv * gw[e * Hn + h];
  }
#pragma unroll
  for (int o = 16; o > 0; o >>= 1) {
#pragma unroll
    for (int e = 0; e < En; ++e) acc[e] += __shfl_down(acc[e], o, 32);
  }
  if (lane == 0) {
    float mx = acc[0];
#pragma unroll
    for (int e = 1; e < En; ++e) mx = fmaxf(mx, acc[e]);
    float p[En];
#pragma unroll
    for (int e = 0; e < En; ++e) p[e] = __expf(acc[e] - mx);
    int i0 = 0;
#pragma unroll
    for (int e = 1; e < En; ++e) if (p[e] > p[i0]) i0 = e;
    int i1 = (i0 == 0) ? 1 : 0;
#pragma unroll
    for (int e = 0; e < En; ++e) if (e != i0 && p[e] > p[i1]) i1 = e;
    float s2 = p[i0] + p[i1];
    float w0 = p[i0] / s2, w1v = p[i1] / s2;
#pragma unroll
    for (int e = 0; e < En; ++e)
      cw[t * En + e] = (e == i0) ? w0 : ((e == i1) ? w1v : 0.0f);
    int p0 = atomicAdd(&cnt[i0], 1); idx[i0 * Tn + p0] = t;
    int p1 = atomicAdd(&cnt[i1], 1); idx[i1 * Tn + p1] = t;
  }
}

// ---------------------------------------------------------------------------
// Kernel 2: exclusive scan of 12 counts -> row offsets into compact h buffer.
// ---------------------------------------------------------------------------
__global__ void prefix_kernel(const int* __restrict__ cnt, int* __restrict__ offs) {
  if (threadIdx.x == 0 && blockIdx.x == 0) {
    int s = 0;
    for (int e = 0; e < En; ++e) { offs[e] = s; s += cnt[e]; }
    offs[En] = s;
  }
}

// ---------------------------------------------------------------------------
// Kernel 3: x fp32 -> bf16 (vectorized: 8 elements / thread).
// ---------------------------------------------------------------------------
__global__ __launch_bounds__(256) void cvt_kernel(
    const float* __restrict__ x, unsigned short* __restrict__ xb) {
  size_t base = ((size_t)blockIdx.x * blockDim.x + threadIdx.x) * 8;
  if (base >= (size_t)Tn * Hn) return;
  const float4* p = (const float4*)(x + base);
  float4 a = p[0], b = p[1];
  uivec4 o;
  o.x = pack_bf16(a.x, a.y); o.y = pack_bf16(a.z, a.w);
  o.z = pack_bf16(b.x, b.y); o.w = pack_bf16(b.z, b.w);
  *(uivec4*)(xb + base) = o;
}

// ---------------------------------------------------------------------------
// Kernel 4: up-projection + SwiGLU (64 tokens x 64 F-cols, K over H by 64).
// Double-buffered pipeline over K.
// ---------------------------------------------------------------------------
__global__ __launch_bounds__(256) void ffn1_kernel(
    const unsigned short* __restrict__ xb, const float* __restrict__ w1,
    const float* __restrict__ w3, const int* __restrict__ idx,
    const int* __restrict__ cnt, const int* __restrict__ offs,
    unsigned short* __restrict__ hb) {
  const int e = blockIdx.z;
  const int count = cnt[e];
  const int m0 = blockIdx.y * 64;
  if (m0 >= count) return;
  const int n0 = blockIdx.x * 64;
  const int ob = offs[e];

  const int tid  = threadIdx.x;
  const int lane = tid & 31;
  const int wv   = tid >> 5;
  const int wm   = wv & 3;   // m-subtile 0..3
  const int wng  = wv >> 2;  // n-group 0..1 (two 16-wide n-subtiles each)

  __shared__ unsigned short a_lds[2][64 * LSTRIDE];
  __shared__ unsigned short b1_lds[2][64 * LSTRIDE];
  __shared__ unsigned short b3_lds[2][64 * LSTRIDE];

  // staging: each thread owns one tile row, two 8-element (16B) chunks
  const int lrow = tid >> 2;
  const int lc   = tid & 3;
  const int c0   = lc * 8;
  const int c1   = (lc + 4) * 8;
  int pos = m0 + lrow;
  if (pos >= count) pos = count - 1;  // clamp; masked on store
  const unsigned short* agp = xb + (size_t)idx[e * Tn + pos] * Hn;
  const float* b1gp = w1 + ((size_t)e * Fn + n0 + lrow) * Hn;
  const float* b3gp = w3 + ((size_t)e * Fn + n0 + lrow) * Hn;

  FragC acc1[2], acc3[2];
#pragma unroll
  for (int ns = 0; ns < 2; ++ns)
#pragma unroll
    for (int j = 0; j < 8; ++j) { acc1[ns].f[j] = 0.0f; acc3[ns].f[j] = 0.0f; }

  const int row16 = lane & 15;
  const int kb    = (lane & 16) ? 8 : 0;

  float4 r1[4], r3[4];  // register-staged weight chunks for the next buffer

  auto issueA = [&](int buf, int k0) {
    async_copy_b128((unsigned int)(size_t)&a_lds[buf][lrow * LSTRIDE + c0],
                    agp + k0 + c0);
    async_copy_b128((unsigned int)(size_t)&a_lds[buf][lrow * LSTRIDE + c1],
                    agp + k0 + c1);
  };
  auto loadB = [&](int k0) {
    const float4* p;
    p = (const float4*)(b1gp + k0 + c0); r1[0] = p[0]; r1[1] = p[1];
    p = (const float4*)(b1gp + k0 + c1); r1[2] = p[0]; r1[3] = p[1];
    p = (const float4*)(b3gp + k0 + c0); r3[0] = p[0]; r3[1] = p[1];
    p = (const float4*)(b3gp + k0 + c1); r3[2] = p[0]; r3[3] = p[1];
  };
  auto storeB = [&](int buf) {
    uivec4 o;
    o.x = pack_bf16(r1[0].x, r1[0].y); o.y = pack_bf16(r1[0].z, r1[0].w);
    o.z = pack_bf16(r1[1].x, r1[1].y); o.w = pack_bf16(r1[1].z, r1[1].w);
    *(uivec4*)&b1_lds[buf][lrow * LSTRIDE + c0] = o;
    o.x = pack_bf16(r1[2].x, r1[2].y); o.y = pack_bf16(r1[2].z, r1[2].w);
    o.z = pack_bf16(r1[3].x, r1[3].y); o.w = pack_bf16(r1[3].z, r1[3].w);
    *(uivec4*)&b1_lds[buf][lrow * LSTRIDE + c1] = o;
    o.x = pack_bf16(r3[0].x, r3[0].y); o.y = pack_bf16(r3[0].z, r3[0].w);
    o.z = pack_bf16(r3[1].x, r3[1].y); o.w = pack_bf16(r3[1].z, r3[1].w);
    *(uivec4*)&b3_lds[buf][lrow * LSTRIDE + c0] = o;
    o.x = pack_bf16(r3[2].x, r3[2].y); o.y = pack_bf16(r3[2].z, r3[2].w);
    o.z = pack_bf16(r3[3].x, r3[3].y); o.w = pack_bf16(r3[3].z, r3[3].w);
    *(uivec4*)&b3_lds[buf][lrow * LSTRIDE + c1] = o;
  };
  auto compute = [&](int buf) {
#pragma unroll
    for (int kc = 0; kc < 64; kc += 32) {
      FragAB A;
      const unsigned short* ap =
          &a_lds[buf][(wm * 16 + row16) * LSTRIDE + kc + kb];
#pragma unroll
      for (int j = 0; j < 4; ++j) {
        A.u[j]     = *(const unsigned int*)(ap + 2 * j);
        A.u[4 + j] = *(const unsigned int*)(ap + 16 + 2 * j);
      }
#pragma unroll
      for (int ns = 0; ns < 2; ++ns) {
        const unsigned short* bp1 =
            &b1_lds[buf][((wng * 2 + ns) * 16 + row16) * LSTRIDE + kc + kb];
        const unsigned short* bp3 =
            &b3_lds[buf][((wng * 2 + ns) * 16 + row16) * LSTRIDE + kc + kb];
        FragAB B1, B3;
#pragma unroll
        for (int j = 0; j < 4; ++j) {
          B1.u[j]     = *(const unsigned int*)(bp1 + 2 * j);
          B1.u[4 + j] = *(const unsigned int*)(bp1 + 16 + 2 * j);
          B3.u[j]     = *(const unsigned int*)(bp3 + 2 * j);
          B3.u[4 + j] = *(const unsigned int*)(bp3 + 16 + 2 * j);
        }
        acc1[ns].v = __builtin_amdgcn_wmma_f32_16x16x32_bf16(
            false, A.v, false, B1.v, (short)0, acc1[ns].v, false, false);
        acc3[ns].v = __builtin_amdgcn_wmma_f32_16x16x32_bf16(
            false, A.v, false, B3.v, (short)0, acc3[ns].v, false, false);
      }
    }
  };

  // pipeline: prologue stages buf0; each iter overlaps next-stage with compute
  const int NK = Hn / 64;
  issueA(0, 0);
  loadB(0);
  storeB(0);
  async_wait0();
  __syncthreads();
  for (int i = 0; i < NK - 1; ++i) {
    const int buf = i & 1, nb = buf ^ 1;
    const int k0n = (i + 1) * 64;
    issueA(nb, k0n);   // async A(next): runs during compute
    loadB(k0n);        // B(next) global loads in flight during compute
    compute(buf);
    storeB(nb);        // convert + LDS store after WMMAs
    async_wait0();
    __syncthreads();
  }
  compute((NK - 1) & 1);

  // epilogue: SwiGLU fuse + bf16 store into compacted h buffer
  const int mh = (lane & 16) >> 1;
#pragma unroll
  for (int ns = 0; ns < 2; ++ns) {
    int n = n0 + (wng * 2 + ns) * 16 + row16;
#pragma unroll
    for (int r = 0; r < 8; ++r) {
      int mi = m0 + wm * 16 + r + mh;
      if (mi < count) {
        float g  = acc1[ns].f[r];
        float hv = g * __builtin_amdgcn_rcpf(1.0f + __expf(-g)) * acc3[ns].f[r];
        hb[(size_t)(ob + mi) * Fn + n] = f2bf_fast(hv);
      }
    }
  }
}

// ---------------------------------------------------------------------------
// Kernel 5: down-projection + weighted scatter-add.
// ---------------------------------------------------------------------------
__global__ __launch_bounds__(256) void ffn2_kernel(
    const unsigned short* __restrict__ hb, const float* __restrict__ w2,
    const int* __restrict__ idx, const int* __restrict__ cnt,
    const int* __restrict__ offs, const float* __restrict__ cw,
    float* __restrict__ out) {
  const int e = blockIdx.z;
  const int count = cnt[e];
  const int m0 = blockIdx.y * 64;
  if (m0 >= count) return;
  const int n0 = blockIdx.x * 64;
  const int ob = offs[e];

  const int tid  = threadIdx.x;
  const int lane = tid & 31;
  const int wv   = tid >> 5;
  const int wm   = wv & 3;
  const int wng  = wv >> 2;

  __shared__ unsigned short a_lds[2][64 * LSTRIDE];
  __shared__ unsigned short b_lds[2][64 * LSTRIDE];

  const int lrow = tid >> 2;
  const int lc   = tid & 3;
  const int c0   = lc * 8;
  const int c1   = (lc + 4) * 8;
  int pos = m0 + lrow;
  if (pos >= count) pos = count - 1;
  const unsigned short* agp = hb + (size_t)(ob + pos) * Fn;
  const float* bgp = w2 + ((size_t)e * Hn + n0 + lrow) * Fn;

  FragC acc[2];
#pragma unroll
  for (int ns = 0; ns < 2; ++ns)
#pragma unroll
    for (int j = 0; j < 8; ++j) acc[ns].f[j] = 0.0f;

  const int row16 = lane & 15;
  const int kb    = (lane & 16) ? 8 : 0;

  float4 rb[4];

  auto issueA = [&](int buf, int k0) {
    async_copy_b128((unsigned int)(size_t)&a_lds[buf][lrow * LSTRIDE + c0],
                    agp + k0 + c0);
    async_copy_b128((unsigned int)(size_t)&a_lds[buf][lrow * LSTRIDE + c1],
                    agp + k0 + c1);
  };
  auto loadB = [&](int k0) {
    const float4* p;
    p = (const float4*)(bgp + k0 + c0); rb[0] = p[0]; rb[1] = p[1];
    p = (const float4*)(bgp + k0 + c1); rb[2] = p[0]; rb[3] = p[1];
  };
  auto storeB = [&](int buf) {
    uivec4 o;
    o.x = pack_bf16(rb[0].x, rb[0].y); o.y = pack_bf16(rb[0].z, rb[0].w);
    o.z = pack_bf16(rb[1].x, rb[1].y); o.w = pack_bf16(rb[1].z, rb[1].w);
    *(uivec4*)&b_lds[buf][lrow * LSTRIDE + c0] = o;
    o.x = pack_bf16(rb[2].x, rb[2].y); o.y = pack_bf16(rb[2].z, rb[2].w);
    o.z = pack_bf16(rb[3].x, rb[3].y); o.w = pack_bf16(rb[3].z, rb[3].w);
    *(uivec4*)&b_lds[buf][lrow * LSTRIDE + c1] = o;
  };
  auto compute = [&](int buf) {
#pragma unroll
    for (int kc = 0; kc < 64; kc += 32) {
      FragAB A;
      const unsigned short* ap =
          &a_lds[buf][(wm * 16 + row16) * LSTRIDE + kc + kb];
#pragma unroll
      for (int j = 0; j < 4; ++j) {
        A.u[j]     = *(const unsigned int*)(ap + 2 * j);
        A.u[4 + j] = *(const unsigned int*)(ap + 16 + 2 * j);
      }
#pragma unroll
      for (int ns = 0; ns < 2; ++ns) {
        const unsigned short* bp =
            &b_lds[buf][((wng * 2 + ns) * 16 + row16) * LSTRIDE + kc + kb];
        FragAB B;
#pragma unroll
        for (int j = 0; j < 4; ++j) {
          B.u[j]     = *(const unsigned int*)(bp + 2 * j);
          B.u[4 + j] = *(const unsigned int*)(bp + 16 + 2 * j);
        }
        acc[ns].v = __builtin_amdgcn_wmma_f32_16x16x32_bf16(
            false, A.v, false, B.v, (short)0, acc[ns].v, false, false);
      }
    }
  };

  const int NK = Fn / 64;
  issueA(0, 0);
  loadB(0);
  storeB(0);
  async_wait0();
  __syncthreads();
  for (int i = 0; i < NK - 1; ++i) {
    const int buf = i & 1, nb = buf ^ 1;
    const int k0n = (i + 1) * 64;
    issueA(nb, k0n);
    loadB(k0n);
    compute(buf);
    storeB(nb);
    async_wait0();
    __syncthreads();
  }
  compute((NK - 1) & 1);

  const int mh = (lane & 16) >> 1;
#pragma unroll
  for (int ns = 0; ns < 2; ++ns) {
    int n = n0 + (wng * 2 + ns) * 16 + row16;
#pragma unroll
    for (int r = 0; r < 8; ++r) {
      int pp = m0 + wm * 16 + r + mh;
      if (pp < count) {
        int t = idx[e * Tn + pp];
        float v = acc[ns].f[r] * cw[t * En + e];
        atomicAdd(&out[(size_t)t * Hn + n], v);
      }
    }
  }
}

// ---------------------------------------------------------------------------
// Launch. Inputs (setup_inputs order): hidden_states, gate_w, w1, w2, w3.
// ---------------------------------------------------------------------------
extern "C" void kernel_launch(void* const* d_in, const int* in_sizes, int n_in,
                              void* d_out, int out_size, void* d_ws, size_t ws_size,
                              hipStream_t stream) {
  const float* x  = (const float*)d_in[0];
  const float* gw = (const float*)d_in[1];
  const float* w1 = (const float*)d_in[2];
  const float* w2 = (const float*)d_in[3];
  const float* w3 = (const float*)d_in[4];
  float* out = (float*)d_out;
  char* ws = (char*)d_ws;

  const size_t off_cw   = 0;                              // float[T*E]
  const size_t off_cnt  = off_cw + (size_t)Tn * En * 4;   // int[16]
  const size_t off_offs = off_cnt + 64;                   // int[16]
  const size_t off_idx  = off_offs + 64;                  // int[E*T]
  const size_t off_xb   = off_idx + (size_t)En * Tn * 4;  // bf16[T*H]
  const size_t off_hb   = off_xb + (size_t)Tn * Hn * 2;   // bf16[2T*F]

  float*          cw_p  = (float*)(ws + off_cw);
  int*            cnt_p = (int*)(ws + off_cnt);
  int*            off_p = (int*)(ws + off_offs);
  int*            idx_p = (int*)(ws + off_idx);
  unsigned short* xb_p  = (unsigned short*)(ws + off_xb);
  unsigned short* hb_p  = (unsigned short*)(ws + off_hb);

  hipMemsetAsync(cnt_p, 0, 64, stream);
  hipMemsetAsync(out, 0, (size_t)out_size * sizeof(float), stream);

  gate_kernel<<<Tn / 8, 256, 0, stream>>>(x, gw, cw_p, cnt_p, idx_p);
  prefix_kernel<<<1, 32, 0, stream>>>(cnt_p, off_p);
  cvt_kernel<<<(Tn * Hn) / (256 * 8), 256, 0, stream>>>(x, xb_p);
  ffn1_kernel<<<dim3(Fn / 64, Tn / 64, En), 256, 0, stream>>>(
      xb_p, w1, w3, idx_p, cnt_p, off_p, hb_p);
  ffn2_kernel<<<dim3(Hn / 64, Tn / 64, En), 256, 0, stream>>>(
      hb_p, w2, idx_p, cnt_p, off_p, cw_p, out);
}
